// MCM_37031208026850
// MI455X (gfx1250) — matrix operand order repaired and checked by reading, not compiled, not run on hardware
//
#include <hip/hip_runtime.h>

#define DEV __device__ __forceinline__

typedef __attribute__((ext_vector_type(16))) __bf16 v16bf;
typedef __attribute__((ext_vector_type(8)))  float  v8f;

union FragU { uint4 q[2]; v16bf v; };

DEV unsigned short f32_to_bf16_bits(float f) {
  unsigned int u = __float_as_uint(f);
  u += 0x7FFFu + ((u >> 16) & 1u);           // round-to-nearest-even
  return (unsigned short)(u >> 16);
}

// A-matrix 16x32 bf16 fragment (ISA 7.12.2): lane holds row m0+(lane&15);
// chunks of 8 at K = k0+8*hi and K = k0+16+8*hi.
DEV v16bf load_frag_a(const unsigned short* __restrict__ A, int lda, int m0, int k0, int lane) {
  const int l = lane & 15, hi = lane >> 4;
  const unsigned short* p = A + (size_t)(m0 + l) * lda + (k0 + 8 * hi);
  FragU f;
  f.q[0] = *(const uint4*)(p);
  f.q[1] = *(const uint4*)(p + 16);
  return f.v;
}

// B-matrix 32x16 bf16 fragment where math-B = Bmat^T (Bmat is (N,K) row-major):
// lane holds Bmat row n0+(lane&15); chunks of 8 at K = k0+8*hi and K = k0+16+8*hi.
DEV v16bf load_frag_b(const unsigned short* __restrict__ B, int ldb, int n0, int k0, int lane) {
  const int n = lane & 15, hi = lane >> 4;
  const unsigned short* p = B + (size_t)(n0 + n) * ldb + (k0 + 8 * hi);
  FragU f;
  f.q[0] = *(const uint4*)(p);
  f.q[1] = *(const uint4*)(p + 16);
  return f.v;
}

// Out[m,n] = sum_k A[m,k]*B[n,k] + bias[n]; A:(M,K) bf16 rm, B:(N,K) bf16 rm.
// Each wave computes a 16x64 output slab (4 N-tiles) so every A fragment feeds
// 4 independent WMMA accumulation chains.
template <bool OUT_F32>
__global__ __launch_bounds__(32) void gemm_abt_kernel(
    const unsigned short* __restrict__ A, const unsigned short* __restrict__ B,
    const float* __restrict__ bias, void* __restrict__ out, int N, int K) {
  const int lane = threadIdx.x & 31;
  const int m0 = blockIdx.x * 16;
  const int n0 = blockIdx.y * 64;
  v8f acc[4] = {};
  for (int k0 = 0; k0 < K; k0 += 32) {
    v16bf a = load_frag_a(A, K, m0, k0, lane);
#pragma unroll
    for (int t = 0; t < 4; ++t) {
      v16bf b = load_frag_b(B, K, n0 + 16 * t, k0, lane);
      acc[t] = __builtin_amdgcn_wmma_f32_16x16x32_bf16(false, a, false, b, (short)0, acc[t], false, false);
    }
  }
  const int n = lane & 15, hi = lane >> 4;
#pragma unroll
  for (int t = 0; t < 4; ++t) {
    const int nn = n0 + 16 * t + n;
    const float bs = bias ? bias[nn] : 0.0f;
#pragma unroll
    for (int r = 0; r < 8; ++r) {
      const int m = m0 + r + 8 * hi;
      const float v = acc[t][r] + bs;
      if (OUT_F32) ((float*)out)[(size_t)m * N + nn] = v;
      else ((unsigned short*)out)[(size_t)m * N + nn] = f32_to_bf16_bits(v);
    }
  }
}

// Streaming pool_gate row statistics: co[i] = sum_b2 max_{j in b2} S[i,j] + sum_j S[i,j]/1024
// with S = Q*K^T (4096x4096, K=512). One wave per 16-row strip.
// Reductions are hoisted: per-lane accumulation in the hot loop (co-executes with WMMA),
// one cross-lane max butterfly per 1024-col block, one cross-lane sum butterfly per strip.
__global__ __launch_bounds__(32) void gate_rowstats_kernel(
    const unsigned short* __restrict__ Q, const unsigned short* __restrict__ Kmat,
    float* __restrict__ co) {
  const int lane = threadIdx.x & 31;
  const int m0 = blockIdx.x * 16;
  v16bf aQ[16];
#pragma unroll
  for (int kt = 0; kt < 16; ++kt) aQ[kt] = load_frag_a(Q, 512, m0, kt * 32, lane);

  float rowsum[8], comax[8];
#pragma unroll
  for (int r = 0; r < 8; ++r) { rowsum[r] = 0.f; comax[r] = 0.f; }

  for (int blk = 0; blk < 4; ++blk) {
    float blkmax[8];
#pragma unroll
    for (int r = 0; r < 8; ++r) blkmax[r] = -3.0e38f;

    for (int jt2 = 0; jt2 < 32; ++jt2) {        // two 16-col tiles per iteration
      const int j0 = blk * 1024 + jt2 * 32;
      if (jt2 < 31) {                           // prefetch next tile pair's rows
        const int n = lane & 15;
        __builtin_prefetch(Kmat + (size_t)(j0 + 32 + n) * 512, 0, 1);
        __builtin_prefetch(Kmat + (size_t)(j0 + 48 + n) * 512, 0, 1);
      }
      v8f acc0 = {}, acc1 = {};
#pragma unroll
      for (int kt = 0; kt < 16; ++kt) {
        v16bf b0 = load_frag_b(Kmat, 512, j0,      kt * 32, lane);
        v16bf b1 = load_frag_b(Kmat, 512, j0 + 16, kt * 32, lane);
        acc0 = __builtin_amdgcn_wmma_f32_16x16x32_bf16(false, aQ[kt], false, b0, (short)0, acc0, false, false);
        acc1 = __builtin_amdgcn_wmma_f32_16x16x32_bf16(false, aQ[kt], false, b1, (short)0, acc1, false, false);
      }
#pragma unroll
      for (int r = 0; r < 8; ++r) {
        rowsum[r] += acc0[r] + acc1[r];
        blkmax[r] = fmaxf(blkmax[r], fmaxf(acc0[r], acc1[r]));
      }
    }
    // one cross-lane max butterfly per block (cols of a tile live in 16 lanes)
#pragma unroll
    for (int off = 1; off < 16; off <<= 1) {
#pragma unroll
      for (int r = 0; r < 8; ++r)
        blkmax[r] = fmaxf(blkmax[r], __shfl_xor(blkmax[r], off, 32));
    }
#pragma unroll
    for (int r = 0; r < 8; ++r) comax[r] += blkmax[r];
  }
  // one cross-lane sum butterfly per strip
#pragma unroll
  for (int off = 1; off < 16; off <<= 1) {
#pragma unroll
    for (int r = 0; r < 8; ++r) rowsum[r] += __shfl_xor(rowsum[r], off, 32);
  }
  if ((lane & 15) == 0) {
    const int hi = lane >> 4;
#pragma unroll
    for (int r = 0; r < 8; ++r)
      co[m0 + r + 8 * hi] = comax[r] + rowsum[r] * (1.0f / 1024.0f);
  }
}

__global__ void softmax1024_kernel(const float* __restrict__ co, float* __restrict__ gate) {
  __shared__ float sm[8];
  const float* x = co + (size_t)blockIdx.x * 1024;
  float* y = gate + (size_t)blockIdx.x * 1024;
  const int t = threadIdx.x;
  const float SCALE = 0.04419417382415922f;  // 1/sqrt(512)
  float v[4];
  float mx = -3.0e38f;
#pragma unroll
  for (int i = 0; i < 4; ++i) { v[i] = x[t + 256 * i] * SCALE; mx = fmaxf(mx, v[i]); }
#pragma unroll
  for (int off = 16; off > 0; off >>= 1) mx = fmaxf(mx, __shfl_xor(mx, off, 32));
  if ((t & 31) == 0) sm[t >> 5] = mx;
  __syncthreads();
  mx = sm[0];
#pragma unroll
  for (int i = 1; i < 8; ++i) mx = fmaxf(mx, sm[i]);
  float s = 0.f;
#pragma unroll
  for (int i = 0; i < 4; ++i) { v[i] = __expf(v[i] - mx); s += v[i]; }
#pragma unroll
  for (int off = 16; off > 0; off >>= 1) s += __shfl_xor(s, off, 32);
  __syncthreads();
  if ((t & 31) == 0) sm[t >> 5] = s;
  __syncthreads();
  s = 0.f;
#pragma unroll
  for (int i = 0; i < 8; ++i) s += sm[i];
  const float inv = 1.0f / s;
#pragma unroll
  for (int i = 0; i < 4; ++i) y[t + 256 * i] = v[i] * inv;
}

__global__ void convert_bf16_kernel(const float* __restrict__ in, unsigned short* __restrict__ out, int n) {
  int i = blockIdx.x * 256 + threadIdx.x;
  if (i < n) out[i] = f32_to_bf16_bits(in[i]);
}

// xc (4,512,32,32) -> XCt (4096,512) bf16, XCt[b*1024+p, c] = xc[b,c,p]
__global__ void xc_transpose_bf16_kernel(const float* __restrict__ xc, unsigned short* __restrict__ out) {
  int idx = blockIdx.x * 256 + threadIdx.x;   // (b,c,p)
  int p = idx & 1023, c = (idx >> 10) & 511, b = idx >> 19;
  out[((size_t)(b * 1024 + p)) * 512 + c] = f32_to_bf16_bits(xc[idx]);
}

// Wf[o64,c] = sum_o W64[o64,o]*Wv[o,c]; bfused[o64] = sum_o W64[o64,o]*bv[o]
__global__ void fuse_w_kernel(const float* __restrict__ W64, const float* __restrict__ Wv,
                              const float* __restrict__ bv, unsigned short* __restrict__ Wf,
                              float* __restrict__ bfused) {
  int idx = blockIdx.x * 256 + threadIdx.x;   // 64*512
  int c = idx & 511, o64 = idx >> 9;
  float s = 0.f;
  for (int o = 0; o < 512; ++o) s += W64[o64 * 512 + o] * Wv[o * 512 + c];
  Wf[idx] = f32_to_bf16_bits(s);
  if (c == 0) {
    float sb = 0.f;
    for (int o = 0; o < 512; ++o) sb += W64[o64 * 512 + o] * bv[o];
    bfused[o64] = sb;
  }
}

// x[b,c,p] = gate{A|B}[b,p] * V64[b*1024+p, c&63] + b64[c&63], c in [0,128)
__global__ void build_x_kernel(const float* __restrict__ gateA, const float* __restrict__ gateB,
                               const float* __restrict__ V64, const float* __restrict__ b64,
                               float* __restrict__ x) {
  int idx = blockIdx.x * 256 + threadIdx.x;   // 4*128*1024
  int p = idx & 1023, c = (idx >> 10) & 127, b = idx >> 17;
  int i = b * 1024 + p;
  int cc = c & 63;
  float g = (c < 64 ? gateA : gateB)[b * 1024 + p];
  x[idx] = g * V64[(size_t)i * 64 + cc] + b64[cc];
}

// Direct conv3x3 SAME + relu; in (4,128,32,32), w (64,128,3,3); out channel-offset write.
__global__ void conv3x3_relu_kernel(const float* __restrict__ x, const float* __restrict__ w,
                                    const float* __restrict__ bias, float* __restrict__ out,
                                    int out_ctotal, int out_coff) {
  int idx = blockIdx.x * 256 + threadIdx.x;   // 4*64*1024
  int p = idx & 1023, o = (idx >> 10) & 63, b = idx >> 16;
  int yy = p >> 5, xx = p & 31;
  float acc = bias[o];
  for (int c = 0; c < 128; ++c) {
    const float* xp = x + ((size_t)b * 128 + c) * 1024;
    const float* wp = w + ((size_t)o * 128 + c) * 9;
#pragma unroll
    for (int dy = 0; dy < 3; ++dy) {
      int iy = yy + dy - 1;
      if (iy < 0 || iy > 31) continue;
#pragma unroll
      for (int dx = 0; dx < 3; ++dx) {
        int ix = xx + dx - 1;
        if (ix < 0 || ix > 31) continue;
        acc += xp[iy * 32 + ix] * wp[dy * 3 + dx];
      }
    }
  }
  out[((size_t)b * out_ctotal + out_coff + o) * 1024 + p] = fmaxf(acc, 0.f);
}

extern "C" void kernel_launch(void* const* d_in, const int* in_sizes, int n_in,
                              void* d_out, int out_size, void* d_ws, size_t ws_size,
                              hipStream_t stream) {
  (void)in_sizes; (void)n_in; (void)out_size; (void)ws_size;
  const float* xc   = (const float*)d_in[0];
  const float* xt   = (const float*)d_in[1];
  const float* Wq_c = (const float*)d_in[2];
  const float* bq_c = (const float*)d_in[3];
  const float* Wk_c = (const float*)d_in[4];
  const float* bk_c = (const float*)d_in[5];
  const float* Wv_c = (const float*)d_in[6];
  const float* bv_c = (const float*)d_in[7];
  const float* Wq_t = (const float*)d_in[8];
  const float* bq_t = (const float*)d_in[9];
  const float* Wk_t = (const float*)d_in[10];
  const float* bk_t = (const float*)d_in[11];
  const float* W64  = (const float*)d_in[12];
  const float* b64  = (const float*)d_in[13];
  const float* W1 = (const float*)d_in[14];
  const float* b1 = (const float*)d_in[15];
  const float* W2 = (const float*)d_in[16];
  const float* b2 = (const float*)d_in[17];
  const float* W3 = (const float*)d_in[18];
  const float* b3 = (const float*)d_in[19];
  float* out = (float*)d_out;

  char* ws = (char*)d_ws;
  size_t off = 0;
  auto alloc = [&](size_t bytes) -> char* {
    char* p = ws + off;
    off += (bytes + 255) & ~(size_t)255;
    return p;
  };
  unsigned short* XCt  = (unsigned short*)alloc(4096u * 512 * 2);
  unsigned short* XT   = (unsigned short*)alloc(4096u * 512 * 2);
  unsigned short* Wqc  = (unsigned short*)alloc(512u * 512 * 2);
  unsigned short* Wkc  = (unsigned short*)alloc(512u * 512 * 2);
  unsigned short* Wqt  = (unsigned short*)alloc(512u * 512 * 2);
  unsigned short* Wkt  = (unsigned short*)alloc(512u * 512 * 2);
  unsigned short* W64b = (unsigned short*)alloc(64u * 512 * 2);
  unsigned short* Wfc  = (unsigned short*)alloc(64u * 512 * 2);
  float* bfused = (float*)alloc(64 * 4);
  unsigned short* CQ = (unsigned short*)alloc(4096u * 512 * 2);
  unsigned short* CK = (unsigned short*)alloc(4096u * 512 * 2);
  unsigned short* TQ = (unsigned short*)alloc(4096u * 512 * 2);
  unsigned short* TK = (unsigned short*)alloc(4096u * 512 * 2);
  float* V64c = (float*)alloc(4096u * 64 * 4);
  float* V64t = (float*)alloc(4096u * 64 * 4);
  float* co   = (float*)alloc(4u * 4096 * 4);
  float* gate = (float*)alloc(4u * 4096 * 4);
  float* x1 = (float*)alloc(4u * 128 * 1024 * 4);
  float* x2 = (float*)alloc(4u * 128 * 1024 * 4);
  float* x3 = (float*)alloc(4u * 128 * 1024 * 4);

  // --- stage 0: layout/precision prep ---
  xc_transpose_bf16_kernel<<<8192, 256, 0, stream>>>(xc, XCt);
  convert_bf16_kernel<<<8192, 256, 0, stream>>>(xt, XT, 4096 * 512);
  convert_bf16_kernel<<<1024, 256, 0, stream>>>(Wq_c, Wqc, 512 * 512);
  convert_bf16_kernel<<<1024, 256, 0, stream>>>(Wk_c, Wkc, 512 * 512);
  convert_bf16_kernel<<<1024, 256, 0, stream>>>(Wq_t, Wqt, 512 * 512);
  convert_bf16_kernel<<<1024, 256, 0, stream>>>(Wk_t, Wkt, 512 * 512);
  convert_bf16_kernel<<<128, 256, 0, stream>>>(W64, W64b, 64 * 512);
  fuse_w_kernel<<<128, 256, 0, stream>>>(W64, Wv_c, bv_c, Wfc, bfused);

  // --- stage 1: projections (WMMA bf16), 16x64 slab per wave ---
  dim3 g512(256, 8), g64(256, 1);
  gemm_abt_kernel<false><<<g512, 32, 0, stream>>>(XCt, Wqc, bq_c, CQ, 512, 512);
  gemm_abt_kernel<false><<<g512, 32, 0, stream>>>(XCt, Wkc, bk_c, CK, 512, 512);
  gemm_abt_kernel<false><<<g512, 32, 0, stream>>>(XT,  Wqt, bq_t, TQ, 512, 512);
  gemm_abt_kernel<false><<<g512, 32, 0, stream>>>(XT,  Wkt, bk_t, TK, 512, 512);
  // value paths with the 512->64 conv1x1 folded in (gate commutes past it)
  gemm_abt_kernel<true><<<g64, 32, 0, stream>>>(XCt, Wfc,  bfused,  V64c, 64, 512);
  gemm_abt_kernel<true><<<g64, 32, 0, stream>>>(XT,  W64b, nullptr, V64t, 64, 512);

  // --- stage 2: streaming pooled-score stats (WMMA, no 4096x4096 materialization) ---
  gate_rowstats_kernel<<<256, 32, 0, stream>>>(CQ, CK, co + 0);       // c_gate
  gate_rowstats_kernel<<<256, 32, 0, stream>>>(TQ, TK, co + 4096);    // t_gate
  gate_rowstats_kernel<<<256, 32, 0, stream>>>(CQ, TK, co + 8192);    // ct_gate
  gate_rowstats_kernel<<<256, 32, 0, stream>>>(TQ, CK, co + 12288);   // tc_gate
  softmax1024_kernel<<<16, 256, 0, stream>>>(co, gate);

  // --- stage 3: gated 128-channel fusion inputs ---
  build_x_kernel<<<2048, 256, 0, stream>>>(gate + 0,    gate + 8192,  V64c, b64, x1);
  build_x_kernel<<<2048, 256, 0, stream>>>(gate + 4096, gate + 12288, V64t, b64, x2);

  // --- stage 4: fusion convs ---
  conv3x3_relu_kernel<<<1024, 256, 0, stream>>>(x1, W1, b1, x3, 128, 0);   // c_ct
  conv3x3_relu_kernel<<<1024, 256, 0, stream>>>(x2, W2, b2, x3, 128, 64);  // t_ct
  conv3x3_relu_kernel<<<1024, 256, 0, stream>>>(x3, W3, b3, out, 64, 0);   // out
}
